// RNNScanList_12489764897122
// MI455X (gfx1250) — matrix operand order repaired and checked by reading, not compiled, not run on hardware
//
#include <hip/hip_runtime.h>

typedef __attribute__((ext_vector_type(8)))  float          v8f;
typedef __attribute__((ext_vector_type(16))) __bf16         v16bf;
typedef __attribute__((ext_vector_type(8)))  __bf16         bf16x8;
typedef __attribute__((ext_vector_type(8)))  unsigned short ushort8;
typedef __attribute__((ext_vector_type(4)))  float          float4v;

constexpr int LYR = 3;
constexpr int BB  = 16;
constexpr int TT  = 4096;
constexpr int DD  = 512;
constexpr int K2  = 1024;            // 2*D (cat dimension)
constexpr int PITCH = DD + 8;        // LDS pitch: 1040B = 65*16B -> conflict-free rows

union BF8 { bf16x8 b; ushort8 u; };
union BF1 { __bf16 b; unsigned short u; };

// hardware bf16 convert (v_cvt_*bf16_f32) instead of manual RNE bit-twiddling
__device__ __forceinline__ unsigned short f2bf(float f) {
  BF1 v; v.b = (__bf16)f; return v.u;
}
__device__ __forceinline__ ushort8 cvt8(float4v f0, float4v f1) {
  BF8 r;
  r.b[0]=(__bf16)f0[0]; r.b[1]=(__bf16)f0[1]; r.b[2]=(__bf16)f0[2]; r.b[3]=(__bf16)f0[3];
  r.b[4]=(__bf16)f1[0]; r.b[5]=(__bf16)f1[1]; r.b[6]=(__bf16)f1[2]; r.b[7]=(__bf16)f1[3];
  return r.u;
}

union ABu { ushort8 u[2]; v16bf v; };

// ---------------- parallel weight conversion: fp32 -> bf16, row-major [L][D][2D]
__global__ void RNNScan_wconv(const float* __restrict__ W,
                              unsigned short* __restrict__ Wb, int n8) {
  int i = blockIdx.x * blockDim.x + threadIdx.x;
  int stride = gridDim.x * blockDim.x;
  const float4v* src = (const float4v*)W;
  ushort8* dst = (ushort8*)Wb;
  for (; i < n8; i += stride) dst[i] = cvt8(src[2 * i], src[2 * i + 1]);
}

// ---------------- persistent single-workgroup scan: 32 waves, one N-tile each
__global__ void __launch_bounds__(1024)
RNNScan_scan(const float* __restrict__ initial,      // [L, B, D]
             const float* __restrict__ x,            // [B, T, D]
             const unsigned short* __restrict__ Wbf, // [L, D, 2D] bf16
             const float* __restrict__ bias,         // [L, D]
             float* __restrict__ out) {              // [B, T, D]
  __shared__ unsigned short hbuf[LYR][BB][PITCH];    // bf16 hidden mirrors (A source)
  __shared__ unsigned short xbuf[2][BB][PITCH];      // double-buffered bf16 x_t tile

  const int tid   = threadIdx.x;
  const int lane  = tid & 31;
  const int wave  = tid >> 5;              // 0..31 == N tile index
  const int ncol  = lane & 15;             // A row index (M) AND C column (N)
  const int nglob = wave * 16 + ncol;      // global output column
  const int mhi   = (lane >> 4) * 8;       // C/D: upper lanes hold rows 8..15
  const int koffA = (lane >> 4) * 8;       // A: lanes>=16 hold K {8-15, 24-31}
  const int koffB = (lane >> 4) * 8;       // B: lanes>=16 hold K {8-15, 24-31}

  // staging assignment: thread -> 8 contiguous elems of a [16][512] tile
  const int sm = (tid * 8) >> 9;           // batch row
  const int sd = (tid * 8) & (DD - 1);     // column

  // ---- init: h lives in WMMA accumulator registers for this wave's 16 columns
  v8f   acc[LYR];
  float bval[LYR];
#pragma unroll
  for (int li = 0; li < LYR; ++li) {
    bval[li] = bias[li * DD + nglob];
#pragma unroll
    for (int r = 0; r < 8; ++r)
      acc[li][r] = initial[(li * BB + (r + mhi)) * DD + nglob];
  }
  // ---- init bf16 LDS mirrors of h, and stage x[:, 0, :]
#pragma unroll
  for (int li = 0; li < LYR; ++li) {
    const float4v* s = (const float4v*)&initial[(li * BB + sm) * DD + sd];
    *(ushort8*)&hbuf[li][sm][sd] = cvt8(s[0], s[1]);
  }
  {
    const float4v* s = (const float4v*)&x[((size_t)sm * TT + 0) * DD + sd];
    *(ushort8*)&xbuf[0][sm][sd] = cvt8(s[0], s[1]);
  }
  __syncthreads();

  for (int t = 0; t < TT; ++t) {
    const int cur = t & 1, nxt = cur ^ 1;
    // ---- issue x[t+1] loads now; latency hides behind 3 layers of WMMA
    const int tn = (t + 1 < TT) ? (t + 1) : t;
    const float4v* xs = (const float4v*)&x[((size_t)sm * TT + tn) * DD + sd];
    const float4v p0 = xs[0], p1 = xs[1];

#pragma unroll
    for (int li = 0; li < LYR; ++li) {
      const unsigned short* hsrc = &hbuf[li][0][0];
      const unsigned short* isrc = (li == 0) ? &xbuf[cur][0][0] : &hbuf[li - 1][0][0];
      const unsigned short* wrow = Wbf + ((size_t)li * DD + nglob) * K2;
      v8f c = acc[li];                      // C = h_prev  ->  D = cat@W^T + h_prev
      // K 0..511: h_prev half of cat
#pragma unroll 4
      for (int kt = 0; kt < 16; ++kt) {
        const int kk = kt * 32;
        ABu a, w;
        a.u[0] = *(const ushort8*)&hsrc[ncol * PITCH + kk + koffA];
        a.u[1] = *(const ushort8*)&hsrc[ncol * PITCH + kk + koffA + 16];
        w.u[0] = *(const ushort8*)&wrow[kk + koffB];
        w.u[1] = *(const ushort8*)&wrow[kk + koffB + 16];
        c = __builtin_amdgcn_wmma_f32_16x16x32_bf16(false, a.v, false, w.v,
                                                    (short)0, c, false, false);
      }
      // K 512..1023: layer input half of cat (x_t or h of previous layer)
#pragma unroll 4
      for (int kt = 0; kt < 16; ++kt) {
        const int kk = kt * 32;
        ABu a, w;
        a.u[0] = *(const ushort8*)&isrc[ncol * PITCH + kk + koffA];
        a.u[1] = *(const ushort8*)&isrc[ncol * PITCH + kk + koffA + 16];
        w.u[0] = *(const ushort8*)&wrow[DD + kk + koffB];
        w.u[1] = *(const ushort8*)&wrow[DD + kk + koffB + 16];
        c = __builtin_amdgcn_wmma_f32_16x16x32_bf16(false, a.v, false, w.v,
                                                    (short)0, c, false, false);
      }
#pragma unroll
      for (int r = 0; r < 8; ++r) c[r] += bval[li];   // + bias -> h_curr
      acc[li] = c;
      __syncthreads();                      // all waves done READING hbuf/xbuf[cur]
#pragma unroll
      for (int r = 0; r < 8; ++r)           // publish bf16 h_curr
        hbuf[li][r + mhi][nglob] = f2bf(c[r]);
      if (li == LYR - 1) {
        // commit prefetched x[t+1] into the other buffer (read only after barrier)
        *(ushort8*)&xbuf[nxt][sm][sd] = cvt8(p0, p1);
#pragma unroll
        for (int r = 0; r < 8; ++r)
          out[((size_t)(r + mhi) * TT + t) * DD + nglob] = c[r];
      }
      __syncthreads();
    }
  }
}

extern "C" void kernel_launch(void* const* d_in, const int* in_sizes, int n_in,
                              void* d_out, int out_size, void* d_ws, size_t ws_size,
                              hipStream_t stream) {
  const float* initial = (const float*)d_in[0];   // [3, 16, 512]
  const float* xseq    = (const float*)d_in[1];   // [16, 4096, 512]
  const float* W       = (const float*)d_in[2];   // [3, 512, 1024]
  const float* bias    = (const float*)d_in[3];   // [3, 512]
  float* out           = (float*)d_out;           // [16, 4096, 512]
  unsigned short* Wbf  = (unsigned short*)d_ws;   // 3 MB bf16 weights

  const int n8 = LYR * DD * K2 / 8;
  RNNScan_wconv<<<1024, 256, 0, stream>>>(W, Wbf, n8);
  RNNScan_scan<<<1, 1024, 0, stream>>>(initial, xseq, Wbf, bias, out);
}
